// CoNHD_ADMM_Layer_87282325389900
// MI455X (gfx1250) — matrix operand order repaired and checked by reading, not compiled, not run on hardware
//
#include <hip/hip_runtime.h>
#include <cstddef>
#include <cstdint>

#define D_DIM 128
#define L_DIM 16
#define N_PAIR 262144
#define PD_ELEMS ((size_t)N_PAIR * D_DIM)

// pre-swizzled bf16 weight arena in d_ws:
//   16 encoder matrices (128x128) : 32 frags * 512 elems each = 16384 elems
//   wu (384x128)                  : 96 frags * 512 elems      = 49152 elems
#define ENC_MAT_ELEMS 16384
#define WS_WU_BASE    ((size_t)16 * ENC_MAT_ELEMS)
#define WS_ELEMS      (WS_WU_BASE + 49152)
#define WS_BYTES      (WS_ELEMS * 2)

typedef __attribute__((ext_vector_type(8)))  float   v8f;
typedef __attribute__((ext_vector_type(16))) __bf16  v16bf;
typedef __attribute__((ext_vector_type(2)))  __bf16  v2bf;

struct MabParams {
  const float *wq, *bq, *wk, *bk, *wv, *bv, *wo, *bo, *g0, *b0, *g1, *b1;
};
struct PrepArgs { const float* src[17]; };

// ---- WMMA fragment helpers (CDNA5 16-bit layouts, ISA 7.12.2) --------------
// A 16x32 (MxK): lanes 0-15: M=lane, halves0-7 K=0..7, halves8-15 K=16..23;
//                lanes16-31: M=lane-16, K offset +8.
__device__ __forceinline__ v16bf load_A_frag(const float* src, int ld, int kBase, int lane) {
  const int m  = lane & 15;
  const int k0 = kBase + ((lane < 16) ? 0 : 8);
  const float* p = src + m * ld + k0;
  float4 a0 = *(const float4*)(p + 0);
  float4 a1 = *(const float4*)(p + 4);
  float4 a2 = *(const float4*)(p + 16);
  float4 a3 = *(const float4*)(p + 20);
  v16bf a;
  a[0]=(__bf16)a0.x; a[1]=(__bf16)a0.y; a[2]=(__bf16)a0.z; a[3]=(__bf16)a0.w;
  a[4]=(__bf16)a1.x; a[5]=(__bf16)a1.y; a[6]=(__bf16)a1.z; a[7]=(__bf16)a1.w;
  a[8]=(__bf16)a2.x; a[9]=(__bf16)a2.y; a[10]=(__bf16)a2.z; a[11]=(__bf16)a2.w;
  a[12]=(__bf16)a3.x; a[13]=(__bf16)a3.y; a[14]=(__bf16)a3.z; a[15]=(__bf16)a3.w;
  return a;
}

// A 16x32, only K=0..15 real (softmax matrix S), upper K zero-padded.
__device__ __forceinline__ v16bf load_A_frag_k16(const float* src, int ld, int lane) {
  const int m  = lane & 15;
  const int k0 = (lane < 16) ? 0 : 8;
  const float* p = src + m * ld + k0;
  float4 a0 = *(const float4*)(p + 0);
  float4 a1 = *(const float4*)(p + 4);
  v16bf a;
  a[0]=(__bf16)a0.x; a[1]=(__bf16)a0.y; a[2]=(__bf16)a0.z; a[3]=(__bf16)a0.w;
  a[4]=(__bf16)a1.x; a[5]=(__bf16)a1.y; a[6]=(__bf16)a1.z; a[7]=(__bf16)a1.w;
  #pragma unroll
  for (int i = 8; i < 16; ++i) a[i] = (__bf16)0.0f;
  return a;
}

// pre-swizzled bf16 fragment: one 32B contiguous load per lane
__device__ __forceinline__ v16bf load_frag_sw(const __bf16* base, int fragIdx, int lane) {
  return *(const v16bf*)(base + (((size_t)fragIdx * 32 + lane) * 16));
}

// B 32x16 (KxN) from row-major f32 W[k][n] (fallback path, strided)
__device__ __forceinline__ v16bf load_B_frag(const float* W, int ld, int kBase, int nBase, int lane) {
  const int n  = nBase + (lane & 15);
  const int k0 = kBase + ((lane < 16) ? 0 : 8);
  const float* p = W + (size_t)k0 * ld + n;
  v16bf b;
  #pragma unroll
  for (int i = 0; i < 16; ++i) b[i] = (__bf16)p[(size_t)((i & 7) + ((i & 8) ? 16 : 0)) * ld];
  return b;
}

// B 32x16 where element(k,n) = src[n*ld + k]  (K^T for attention scores)
__device__ __forceinline__ v16bf load_Bt_frag(const float* src, int ld, int lane) {
  const int n  = lane & 15;
  const int k0 = (lane < 16) ? 0 : 8;
  const float* p = src + n * ld + k0;
  float4 b0 = *(const float4*)(p + 0);
  float4 b1 = *(const float4*)(p + 4);
  float4 b2 = *(const float4*)(p + 16);
  float4 b3 = *(const float4*)(p + 20);
  v16bf b;
  b[0]=(__bf16)b0.x; b[1]=(__bf16)b0.y; b[2]=(__bf16)b0.z; b[3]=(__bf16)b0.w;
  b[4]=(__bf16)b1.x; b[5]=(__bf16)b1.y; b[6]=(__bf16)b1.z; b[7]=(__bf16)b1.w;
  b[8]=(__bf16)b2.x; b[9]=(__bf16)b2.y; b[10]=(__bf16)b2.z; b[11]=(__bf16)b2.w;
  b[12]=(__bf16)b3.x; b[13]=(__bf16)b3.y; b[14]=(__bf16)b3.z; b[15]=(__bf16)b3.w;
  return b;
}

// B 32x16 with only K=0..15 real (V, zero-padded K)
__device__ __forceinline__ v16bf load_B_frag_k16(const float* src, int ld, int nBase, int lane) {
  v16bf b;
  {
    const float* p = src + (size_t)((lane < 16) ? 0 : 8) * ld + nBase + (lane & 15);
    #pragma unroll
    for (int i = 0; i < 8; ++i) b[i] = (__bf16)p[(size_t)i * ld];
    #pragma unroll
    for (int i = 8; i < 16; ++i) b[i] = (__bf16)0.0f;
  }
  return b;
}

// store pair (m,c),(m,c+1) of X into bf16 A-fragment-swizzled mirror (c even)
__device__ __forceinline__ void store_Xh_pair(__bf16* xh, int m, int c, float v0, float v1) {
  const int kt = c >> 5;
  const int kk = c & 31;
  const int lane = m + ((kk & 8) ? 16 : 0);
  const int i = (kk & 7) + ((kk & 16) ? 8 : 0);
  v2bf pk; pk[0] = (__bf16)v0; pk[1] = (__bf16)v1;
  *(v2bf*)(xh + ((size_t)(kt * 32 + lane) * 16 + i)) = pk;
}

// ---- block LayerNorm over 16x128 (128 threads), writes f32 + bf16 mirror ---
__device__ __forceinline__ void block_ln(const float* buf, const float* g, const float* b,
                                         float* outb, __bf16* xh, float* red, int t) {
  const int row = t >> 3, seg = t & 7;
  const float* br = buf + row * D_DIM + seg * 16;
  float s = 0.f;
  #pragma unroll
  for (int i = 0; i < 16; ++i) s += br[i];
  red[row * 8 + seg] = s;
  __syncthreads();
  float mean = 0.f;
  #pragma unroll
  for (int j = 0; j < 8; ++j) mean += red[row * 8 + j];
  mean *= (1.0f / 128.0f);
  __syncthreads();
  float vs = 0.f;
  #pragma unroll
  for (int i = 0; i < 16; ++i) { float d = br[i] - mean; vs += d * d; }
  red[row * 8 + seg] = vs;
  __syncthreads();
  float var = 0.f;
  #pragma unroll
  for (int j = 0; j < 8; ++j) var += red[row * 8 + j];
  var *= (1.0f / 128.0f);
  const float inv = rsqrtf(var + 1e-5f);
  float tv[16];
  #pragma unroll
  for (int i = 0; i < 16; ++i) {
    const int c = seg * 16 + i;
    tv[i] = (br[i] - mean) * inv * g[c] + b[c];
    outb[row * D_DIM + c] = tv[i];
  }
  #pragma unroll
  for (int i = 0; i < 16; i += 2) store_Xh_pair(xh, row, seg * 16 + i, tv[i], tv[i + 1]);
  __syncthreads();
}

// ---- prep: swizzle all weights into bf16 B-fragment layout in d_ws ---------
__global__ void __launch_bounds__(256)
prep_kernel(PrepArgs pa, uint16_t* wsRaw) {
  __bf16* ws = (__bf16*)wsRaw;
  const int tid = blockIdx.x * 256 + threadIdx.x;     // (frag, lane) pairs
  if (tid >= (16 * 32 + 96) * 32) return;
  const int lane = tid & 31;
  const int frag = tid >> 5;
  const float* W; int kt, nt; size_t base;
  if (frag < 512) {                                   // encoder matrices
    const int m = frag >> 5, f = frag & 31;
    W = pa.src[m]; kt = f >> 3; nt = f & 7;
    base = (size_t)m * ENC_MAT_ELEMS + (size_t)f * 512;
  } else {                                            // wu (384x128)
    const int f = frag - 512;
    W = pa.src[16]; kt = f >> 3; nt = f & 7;
    base = WS_WU_BASE + (size_t)f * 512;
  }
  const int n  = nt * 16 + (lane & 15);
  const int k0 = kt * 32 + ((lane < 16) ? 0 : 8);
  __bf16* dst = ws + base + (size_t)lane * 16;
  #pragma unroll
  for (int i = 0; i < 16; ++i) dst[i] = (__bf16)W[(size_t)(k0 + (i & 7) + ((i & 8) ? 16 : 0)) * D_DIM + n];
}

// ---- side kernel: async gather + OrderPE + 2x SAB(MAB) + scatter -----------
// 128 threads = 4 waves; wave w owns cols [32w,32w+32) == attention head w.
template <bool EDGE, bool WS>
__global__ void __launch_bounds__(128)
side_kernel(const float* __restrict__ co, const float* __restrict__ msg,
            const float* __restrict__ wgt, const int* __restrict__ idx,
            const float* __restrict__ pew, const float* __restrict__ peb,
            MabParams mp0, MabParams mp1,
            const uint16_t* __restrict__ wsRaw, int matBase0,
            float* __restrict__ out) {
  __shared__ float  sX [L_DIM * D_DIM];
  __shared__ float  sMV[L_DIM * D_DIM];
  __shared__ float  sQ [L_DIM * D_DIM];
  __shared__ float  sK [L_DIM * D_DIM];
  __shared__ float  sV [L_DIM * D_DIM];
  __shared__ float  sO [L_DIM * D_DIM];
  __shared__ float  sS [4 * 16 * 16];
  __shared__ float  sRed[16 * 8];
  __shared__ __bf16 sXh[4 * 32 * 16];   // X in A-fragment swizzle (bf16)

  const __bf16* ws = (const __bf16*)wsRaw;
  const int t = threadIdx.x;
  const int lane = t & 31;
  const int w = t >> 5;          // wave id == head id
  const int g = blockIdx.x;      // group (node / hyperedge)

  if constexpr (WS) {            // warm L2 for this side's swizzled weights
    const __bf16* p = ws + (size_t)matBase0 * ENC_MAT_ELEMS + (size_t)t * 128;
    __builtin_prefetch(p, 0, 1);
    __builtin_prefetch(p + ENC_MAT_ELEMS, 0, 1);
    __builtin_prefetch(p + 2 * ENC_MAT_ELEMS, 0, 1);
    __builtin_prefetch(p + 3 * ENC_MAT_ELEMS, 0, 1);
  } else {
    const size_t off = (size_t)t * 128;
    __builtin_prefetch(mp0.wq + off, 0, 1);
    __builtin_prefetch(mp0.wk + off, 0, 1);
    __builtin_prefetch(mp0.wv + off, 0, 1);
    __builtin_prefetch(mp0.wo + off, 0, 1);
  }

  const int row = t >> 3, seg = t & 7;
  const int grow = EDGE ? idx[g * 16 + row] : (g * 16 + row);

  // ---- async gather: stage co/msg rows into LDS (sQ/sK as temp) ----
  {
    const float* gv = co  + (size_t)grow * D_DIM + seg * 16;
    const float* gm = msg + (size_t)grow * D_DIM + seg * 16;
    const uint32_t lv = (uint32_t)(uintptr_t)(sQ + row * D_DIM + seg * 16);
    const uint32_t lm = (uint32_t)(uintptr_t)(sK + row * D_DIM + seg * 16);
    #pragma unroll
    for (int i = 0; i < 4; ++i) {
      asm volatile("global_load_async_to_lds_b128 %0, %1, off"
                   :: "v"(lv + 16u * i), "v"(gv + 4 * i) : "memory");
      asm volatile("global_load_async_to_lds_b128 %0, %1, off"
                   :: "v"(lm + 16u * i), "v"(gm + 4 * i) : "memory");
    }
    asm volatile("s_wait_asynccnt 0" ::: "memory");
  }
  __syncthreads();

  // ---- OrderPE: X = 2v - m + w@pe_w + pe_b ; MV = m - v ; bf16 mirror ----
  {
    float wl[8];
    #pragma unroll
    for (int j = 0; j < 8; ++j) wl[j] = wgt[(size_t)grow * 8 + j];
    const float* vp = sQ + row * D_DIM + seg * 16;
    const float* mq = sK + row * D_DIM + seg * 16;
    #pragma unroll
    for (int i = 0; i < 16; i += 2) {
      const int c = seg * 16 + i;
      float pe0 = peb[c], pe1 = peb[c + 1];
      #pragma unroll
      for (int j = 0; j < 8; ++j) {
        pe0 += wl[j] * pew[j * D_DIM + c];
        pe1 += wl[j] * pew[j * D_DIM + c + 1];
      }
      const float v0 = vp[i], m0 = mq[i], v1 = vp[i + 1], m1 = mq[i + 1];
      const float x0 = 2.0f * v0 - m0 + pe0, x1 = 2.0f * v1 - m1 + pe1;
      sX[row * D_DIM + c] = x0;  sX[row * D_DIM + c + 1] = x1;
      sMV[row * D_DIM + c] = m0 - v0;  sMV[row * D_DIM + c + 1] = m1 - v1;
      store_Xh_pair(sXh, row, c, x0, x1);
    }
  }
  __syncthreads();

  #pragma unroll 1
  for (int layer = 0; layer < 2; ++layer) {
    const MabParams mp = (layer == 0) ? mp0 : mp1;
    const int matL = matBase0 + layer * 4;

    // hoisted A fragments of X (reused by Q,K,V x 2 n-tiles = 6 WMMAs each)
    v16bf xfr[4];
    #pragma unroll
    for (int kt = 0; kt < 4; ++kt) xfr[kt] = *(const v16bf*)(sXh + (size_t)(kt * 32 + lane) * 16);

    #pragma unroll 1
    for (int pj = 0; pj < 3; ++pj) {
      const float* Wf  = (pj == 0) ? mp.wq : (pj == 1) ? mp.wk : mp.wv;
      const float* bia = (pj == 0) ? mp.bq : (pj == 1) ? mp.bk : mp.bv;
      float* dst       = (pj == 0) ? sQ   : (pj == 1) ? sK   : sV;
      const __bf16* wsM = ws + (size_t)(matL + pj) * ENC_MAT_ELEMS;
      #pragma unroll
      for (int nt = 0; nt < 2; ++nt) {
        const int ntG = w * 2 + nt;
        const int n0  = ntG * 16;
        v8f acc = {};
        #pragma unroll
        for (int kt = 0; kt < 4; ++kt) {
          v16bf b = WS ? load_frag_sw(wsM, kt * 8 + ntG, lane)
                       : load_B_frag(Wf, D_DIM, kt * 32, n0, lane);
          acc = __builtin_amdgcn_wmma_f32_16x16x32_bf16(false, xfr[kt], false, b, (short)0, acc, false, false);
        }
        const int n  = n0 + (lane & 15);
        const int mB = (lane < 16) ? 0 : 8;
        const float bv = bia[n];
        #pragma unroll
        for (int r = 0; r < 8; ++r) dst[(mB + r) * D_DIM + n] = acc[r] + bv;
      }
    }
    __syncthreads();

    // attention head w:  S = softmax(qh @ kh^T / sqrt(D))
    {
      const int hb = w * 32;
      v16bf aq = load_A_frag(sQ, D_DIM, hb, lane);
      v16bf bk = load_Bt_frag(sK + hb, D_DIM, lane);
      v8f s = {};
      s = __builtin_amdgcn_wmma_f32_16x16x32_bf16(false, aq, false, bk, (short)0, s, false, false);
      const float scale = 0.08838834764831845f;  // 1/sqrt(128)
      #pragma unroll
      for (int r = 0; r < 8; ++r) {
        float x = s[r] * scale;
        float mx = x;
        mx = fmaxf(mx, __shfl_xor(mx, 1, 16));
        mx = fmaxf(mx, __shfl_xor(mx, 2, 16));
        mx = fmaxf(mx, __shfl_xor(mx, 4, 16));
        mx = fmaxf(mx, __shfl_xor(mx, 8, 16));
        const float e = __expf(x - mx);
        float sm = e;
        sm += __shfl_xor(sm, 1, 16);
        sm += __shfl_xor(sm, 2, 16);
        sm += __shfl_xor(sm, 4, 16);
        sm += __shfl_xor(sm, 8, 16);
        s[r] = e / sm;
      }
      float* sSw = sS + w * 256;
      const int n  = lane & 15;
      const int mB = (lane < 16) ? 0 : 8;
      #pragma unroll
      for (int r = 0; r < 8; ++r) sSw[(mB + r) * 16 + n] = s[r];
    }
    __syncthreads();

    // O = qh + S @ vh (C preloaded with qh; K zero-padded 16->32)
    {
      const int hb = w * 32;
      v16bf as = load_A_frag_k16(sS + w * 256, 16, lane);
      #pragma unroll
      for (int nt = 0; nt < 2; ++nt) {
        const int n0 = hb + nt * 16;
        const int n  = n0 + (lane & 15);
        const int mB = (lane < 16) ? 0 : 8;
        v8f o;
        #pragma unroll
        for (int r = 0; r < 8; ++r) o[r] = sQ[(mB + r) * D_DIM + n];
        v16bf bv = load_B_frag_k16(sV, D_DIM, n0, lane);
        o = __builtin_amdgcn_wmma_f32_16x16x32_bf16(false, as, false, bv, (short)0, o, false, false);
        #pragma unroll
        for (int r = 0; r < 8; ++r) sO[(mB + r) * D_DIM + n] = o[r];
      }
    }
    __syncthreads();

    block_ln(sO, mp.g0, mp.b0, sX, sXh, sRed, t);     // X = LN0(O), + bf16 mirror

    // rFF: O = X + relu(X @ wo + bo)
    {
      v16bf ffr[4];
      #pragma unroll
      for (int kt = 0; kt < 4; ++kt) ffr[kt] = *(const v16bf*)(sXh + (size_t)(kt * 32 + lane) * 16);
      const __bf16* wsM = ws + (size_t)(matL + 3) * ENC_MAT_ELEMS;
      #pragma unroll
      for (int nt = 0; nt < 2; ++nt) {
        const int ntG = w * 2 + nt;
        const int n0  = ntG * 16;
        v8f acc = {};
        #pragma unroll
        for (int kt = 0; kt < 4; ++kt) {
          v16bf b = WS ? load_frag_sw(wsM, kt * 8 + ntG, lane)
                       : load_B_frag(mp.wo, D_DIM, kt * 32, n0, lane);
          acc = __builtin_amdgcn_wmma_f32_16x16x32_bf16(false, ffr[kt], false, b, (short)0, acc, false, false);
        }
        const int n  = n0 + (lane & 15);
        const int mB = (lane < 16) ? 0 : 8;
        const float bo = mp.bo[n];
        #pragma unroll
        for (int r = 0; r < 8; ++r) {
          const float y = fmaxf(acc[r] + bo, 0.0f);
          sO[(mB + r) * D_DIM + n] = sX[(mB + r) * D_DIM + n] + y;
        }
      }
    }
    __syncthreads();

    block_ln(sO, mp.g1, mp.b1, sX, sXh, sRed, t);     // X = LN1(O), + bf16 mirror
  }

  // scatter: out[idx] = X + (m - v)
  {
    const float4* xr = (const float4*)(sX  + row * D_DIM + seg * 16);
    const float4* mr = (const float4*)(sMV + row * D_DIM + seg * 16);
    float4* orow = (float4*)(out + (size_t)grow * D_DIM + seg * 16);
    #pragma unroll
    for (int i = 0; i < 4; ++i) {
      float4 xv = xr[i], mv = mr[i], ov;
      ov.x = xv.x + mv.x; ov.y = xv.y + mv.y; ov.z = xv.z + mv.z; ov.w = xv.w + mv.w;
      orow[i] = ov;
    }
  }
}

// ---- fusion kernel: co_new = LN([msg_v, msg_e, co0] @ wu + bu) -------------
template <bool WS>
__global__ void __launch_bounds__(128)
fuse_kernel(float* __restrict__ out, const float* __restrict__ c0,
            const float* __restrict__ wu, const float* __restrict__ bu,
            const float* __restrict__ gg, const float* __restrict__ bb,
            const uint16_t* __restrict__ wsRaw) {
  __shared__ float sBuf[64 * D_DIM];
  __shared__ float sRed[64 * 2];
  const __bf16* wsU = (const __bf16*)wsRaw + WS_WU_BASE;
  const int t = threadIdx.x;
  const int lane = t & 31;
  const int w = t >> 5;
  const float* mv = out + PD_ELEMS;       // slice 1 (message_v)
  const float* me = out + 3 * PD_ELEMS;   // slice 3 (message_e)
  const size_t r0 = (size_t)blockIdx.x * 64 + (size_t)w * 16;

  // hoist the 12 activation A-fragments (K = 384 over the concat inputs)
  v16bf afr[12];
  #pragma unroll
  for (int kt = 0; kt < 12; ++kt) {
    const int kb = kt * 32;
    const float* src = (kb < 128) ? (mv + r0 * D_DIM + kb)
                     : (kb < 256) ? (me + r0 * D_DIM + (kb - 128))
                                  : (c0 + r0 * D_DIM + (kb - 256));
    afr[kt] = load_A_frag(src, D_DIM, 0, lane);
  }

  #pragma unroll 1
  for (int nt = 0; nt < 8; ++nt) {
    const int n0 = nt * 16;
    v8f acc = {};
    #pragma unroll
    for (int kt = 0; kt < 12; ++kt) {
      v16bf b = WS ? load_frag_sw(wsU, kt * 8 + nt, lane)
                   : load_B_frag(wu + (size_t)(kt * 32) * D_DIM, D_DIM, 0, n0, lane);
      acc = __builtin_amdgcn_wmma_f32_16x16x32_bf16(false, afr[kt], false, b, (short)0, acc, false, false);
    }
    const int n  = n0 + (lane & 15);
    const int mB = (lane < 16) ? 0 : 8;
    const float bv = bu[n];
    #pragma unroll
    for (int r = 0; r < 8; ++r) sBuf[(w * 16 + mB + r) * D_DIM + n] = acc[r] + bv;
  }
  __syncthreads();

  // LayerNorm (2 threads/row) + write slices 0, 2, copy slice 4
  const int row = t >> 1, half = t & 1;
  const float* br = sBuf + row * D_DIM + half * 64;
  float s = 0.f;
  #pragma unroll
  for (int i = 0; i < 64; ++i) s += br[i];
  sRed[row * 2 + half] = s;
  __syncthreads();
  const float mean = (sRed[row * 2] + sRed[row * 2 + 1]) * (1.0f / 128.0f);
  float vs = 0.f;
  #pragma unroll
  for (int i = 0; i < 64; ++i) { float d = br[i] - mean; vs += d * d; }
  __syncthreads();
  sRed[row * 2 + half] = vs;
  __syncthreads();
  const float var = (sRed[row * 2] + sRed[row * 2 + 1]) * (1.0f / 128.0f);
  const float inv = rsqrtf(var + 1e-5f);
  const size_t grow = (size_t)blockIdx.x * 64 + row;
  float* o0 = out + grow * D_DIM + half * 64;
  float* o2 = out + 2 * PD_ELEMS + grow * D_DIM + half * 64;
  float* o4 = out + 4 * PD_ELEMS + grow * D_DIM + half * 64;
  const float* cr = c0 + grow * D_DIM + half * 64;
  #pragma unroll
  for (int i = 0; i < 64; ++i) {
    const int c = half * 64 + i;
    const float v2 = (br[i] - mean) * inv * gg[c] + bb[c];
    o0[i] = v2; o2[i] = v2; o4[i] = cr[i];
  }
}

extern "C" void kernel_launch(void* const* d_in, const int* in_sizes, int n_in,
                              void* d_out, int out_size, void* d_ws, size_t ws_size,
                              hipStream_t stream) {
  (void)in_sizes; (void)n_in; (void)out_size;
  const float* co_in   = (const float*)d_in[0];
  const float* msg_in  = (const float*)d_in[1];
  const float* co_con  = (const float*)d_in[2];
  const float* msg_con = (const float*)d_in[3];
  const float* co0     = (const float*)d_in[4];
  const float* w_in    = (const float*)d_in[5];
  const float* w_con   = (const float*)d_in[6];
  const int*   idx_e   = (const int*)d_in[7];
  const float* pe_v_w  = (const float*)d_in[8];
  const float* pe_v_b  = (const float*)d_in[9];
  const float* pe_e_w  = (const float*)d_in[10];
  const float* pe_e_b  = (const float*)d_in[11];
  auto mab = [&](int base) {
    MabParams p;
    p.wq = (const float*)d_in[base + 0];  p.bq = (const float*)d_in[base + 1];
    p.wk = (const float*)d_in[base + 2];  p.bk = (const float*)d_in[base + 3];
    p.wv = (const float*)d_in[base + 4];  p.bv = (const float*)d_in[base + 5];
    p.wo = (const float*)d_in[base + 6];  p.bo = (const float*)d_in[base + 7];
    p.g0 = (const float*)d_in[base + 8];  p.b0 = (const float*)d_in[base + 9];
    p.g1 = (const float*)d_in[base + 10]; p.b1 = (const float*)d_in[base + 11];
    return p;
  };
  MabParams ev0 = mab(12), ev1 = mab(24), ee0 = mab(36), ee1 = mab(48);
  const float* wu    = (const float*)d_in[60];
  const float* bu    = (const float*)d_in[61];
  const float* g_ln0 = (const float*)d_in[62];
  const float* b_ln0 = (const float*)d_in[63];
  float* out = (float*)d_out;
  uint16_t* ws = (uint16_t*)d_ws;
  const bool useWS = (ws_size >= (size_t)WS_BYTES);

  if (useWS) {
    PrepArgs pa;
    pa.src[0] = ev0.wq; pa.src[1] = ev0.wk; pa.src[2]  = ev0.wv; pa.src[3]  = ev0.wo;
    pa.src[4] = ev1.wq; pa.src[5] = ev1.wk; pa.src[6]  = ev1.wv; pa.src[7]  = ev1.wo;
    pa.src[8] = ee0.wq; pa.src[9] = ee0.wk; pa.src[10] = ee0.wv; pa.src[11] = ee0.wo;
    pa.src[12]= ee1.wq; pa.src[13]= ee1.wk; pa.src[14] = ee1.wv; pa.src[15] = ee1.wo;
    pa.src[16]= wu;
    prep_kernel<<<76, 256, 0, stream>>>(pa, ws);
    side_kernel<false, true><<<16384, 128, 0, stream>>>(co_in, msg_in, w_in, nullptr,
        pe_v_w, pe_v_b, ev0, ev1, ws, 0, out + PD_ELEMS);
    side_kernel<true, true><<<16384, 128, 0, stream>>>(co_con, msg_con, w_con, idx_e,
        pe_e_w, pe_e_b, ee0, ee1, ws, 8, out + 3 * PD_ELEMS);
    fuse_kernel<true><<<4096, 128, 0, stream>>>(out, co0, wu, bu, g_ln0, b_ln0, ws);
  } else {
    side_kernel<false, false><<<16384, 128, 0, stream>>>(co_in, msg_in, w_in, nullptr,
        pe_v_w, pe_v_b, ev0, ev1, ws, 0, out + PD_ELEMS);
    side_kernel<true, false><<<16384, 128, 0, stream>>>(co_con, msg_con, w_con, idx_e,
        pe_e_w, pe_e_b, ee0, ee1, ws, 8, out + 3 * PD_ELEMS);
    fuse_kernel<false><<<4096, 128, 0, stream>>>(out, co0, wu, bu, g_ln0, b_ln0, ws);
  }
}
